// CorrelationLayer_816043786370
// MI455X (gfx1250) — matrix operand order repaired and checked by reading, not compiled, not run on hardware
//
#include <hip/hip_runtime.h>

typedef __attribute__((ext_vector_type(8)))  _Float16 v8h;
typedef __attribute__((ext_vector_type(16))) _Float16 v16h;
typedef __attribute__((ext_vector_type(8)))  float    v8f;

#define C_DIM 256
#define H_DIM 48
#define W_DIM 64
#define NDISP 21            // 21 displacements per axis, stride 2 in [-20,20]
#define LSTR  264           // LDS row stride in halves (256 + 8 pad, keeps 16B align)
#define GSTR  68            // Gram staging stride in floats

__global__ __launch_bounds__(128)
void corr_wmma_kernel(const float* __restrict__ L,
                      const float* __restrict__ R,
                      float* __restrict__ out)
{
    extern __shared__ unsigned char smem[];
    _Float16* LT = (_Float16*)smem;            // [64][LSTR] : A = L^T (x-major)
    _Float16* RT = LT + 64 * LSTR;             // [64][LSTR] : B   (x'-major)
    float*    G  = (float*)smem;               // [64][GSTR] : reused after barrier

    const int jj = blockIdx.x;                 // y-displacement index 0..20
    const int y  = blockIdx.y;                 // 0..47
    const int b  = blockIdx.z;                 // batch
    const int j  = -20 + 2 * jj;
    const int yp = y - j;
    const int t  = threadIdx.x;                // 0..127

    if (yp < 0 || yp >= H_DIM) {
        // whole output row for this (b, *, jj, y) is zero (r shifted out of range)
        for (int idx = t; idx < NDISP * W_DIM; idx += 128) {
            int ii = idx >> 6, x = idx & 63;
            size_t o = ((size_t)(b * (NDISP * NDISP) + ii * NDISP + jj) * H_DIM + y) * W_DIM + x;
            out[o] = 0.0f;
        }
        return;
    }

    // ---- stage both feature rows into LDS, transposed + f32->f16 ----
    const float* lrow = L + (size_t)b * C_DIM * H_DIM * W_DIM + (size_t)y  * W_DIM;
    const float* rrow = R + (size_t)b * C_DIM * H_DIM * W_DIM + (size_t)yp * W_DIM;
    {
        const int x4 = (t & 15) * 4;           // 0,4,...,60
        const int c0 = t >> 4;                 // 0..7
        for (int c = c0; c < C_DIM; c += 8) {
            float4 lv = *(const float4*)(lrow + (size_t)c * (H_DIM * W_DIM) + x4);
            float4 rv = *(const float4*)(rrow + (size_t)c * (H_DIM * W_DIM) + x4);
            LT[(x4 + 0) * LSTR + c] = (_Float16)lv.x;
            LT[(x4 + 1) * LSTR + c] = (_Float16)lv.y;
            LT[(x4 + 2) * LSTR + c] = (_Float16)lv.z;
            LT[(x4 + 3) * LSTR + c] = (_Float16)lv.w;
            RT[(x4 + 0) * LSTR + c] = (_Float16)rv.x;
            RT[(x4 + 1) * LSTR + c] = (_Float16)rv.y;
            RT[(x4 + 2) * LSTR + c] = (_Float16)rv.z;
            RT[(x4 + 3) * LSTR + c] = (_Float16)rv.w;
        }
    }
    __syncthreads();

    // ---- 64x64 Gram G = L^T * R via v_wmma_f32_16x16x32_f16 ----
    const int lane = t & 31;
    const int wave = t >> 5;
    const int mb = (wave >> 1) * 32;           // wave's 32x32 quadrant
    const int nb = (wave & 1) * 32;

    v8f c00 = {}, c01 = {}, c10 = {}, c11 = {};

    // 16-bit A layout: lanes 0-15 row m, K chunks {0..7,16..23}; lanes 16-31 {8..15,24..31}
    const int am = lane & 15;
    const int ak = (lane >> 4) << 3;           // 0 or 8
    // 16-bit B layout: lane n = lane&15; lanes 0-15 K chunks {0..7,16..23}; lanes 16-31 {8..15,24..31}
    const int bn = lane & 15;
    const int bk = (lane >> 4) << 3;           // 0 or 8

    #pragma unroll
    for (int kk = 0; kk < C_DIM; kk += 32) {
        union { v16h v; v8h h[2]; } a0, a1, b0, b1;
        const _Float16* pa0 = LT + (mb + am) * LSTR + kk + ak;
        a0.h[0] = *(const v8h*)(pa0);          a0.h[1] = *(const v8h*)(pa0 + 16);
        const _Float16* pa1 = pa0 + 16 * LSTR;
        a1.h[0] = *(const v8h*)(pa1);          a1.h[1] = *(const v8h*)(pa1 + 16);
        const _Float16* pb0 = RT + (nb + bn) * LSTR + kk + bk;
        b0.h[0] = *(const v8h*)(pb0);          b0.h[1] = *(const v8h*)(pb0 + 16);
        const _Float16* pb1 = pb0 + 16 * LSTR;
        b1.h[0] = *(const v8h*)(pb1);          b1.h[1] = *(const v8h*)(pb1 + 16);

        c00 = __builtin_amdgcn_wmma_f32_16x16x32_f16(false, a0.v, false, b0.v, (short)0, c00, false, false);
        c01 = __builtin_amdgcn_wmma_f32_16x16x32_f16(false, a0.v, false, b1.v, (short)0, c01, false, false);
        c10 = __builtin_amdgcn_wmma_f32_16x16x32_f16(false, a1.v, false, b0.v, (short)0, c10, false, false);
        c11 = __builtin_amdgcn_wmma_f32_16x16x32_f16(false, a1.v, false, b1.v, (short)0, c11, false, false);
    }

    __syncthreads();   // everyone done reading LT/RT -> safe to alias as G

    // C/D layout: lanes 0-15: VGPR v = row mb+v; lanes 16-31: row mb+8+v; col = nb + (lane&15)
    {
        const int mrow = (lane >> 4) * 8;
        const int ncol = lane & 15;
        #pragma unroll
        for (int v = 0; v < 8; ++v) {
            G[(mb + mrow + v)      * GSTR + (nb + ncol)]      = c00[v];
            G[(mb + mrow + v)      * GSTR + (nb + 16 + ncol)] = c01[v];
            G[(mb + 16 + mrow + v) * GSTR + (nb + ncol)]      = c10[v];
            G[(mb + 16 + mrow + v) * GSTR + (nb + 16 + ncol)] = c11[v];
        }
    }
    __syncthreads();

    // ---- scatter the 21 stride-2 diagonals to the output volume ----
    for (int idx = t; idx < NDISP * W_DIM; idx += 128) {
        int ii = idx >> 6, x = idx & 63;
        int i  = -20 + 2 * ii;
        int xp = x - i;
        float v = (xp >= 0 && xp < W_DIM) ? G[x * GSTR + xp] : 0.0f;
        size_t o = ((size_t)(b * (NDISP * NDISP) + ii * NDISP + jj) * H_DIM + y) * W_DIM + x;
        out[o] = v;
    }
}

extern "C" void kernel_launch(void* const* d_in, const int* in_sizes, int n_in,
                              void* d_out, int out_size, void* d_ws, size_t ws_size,
                              hipStream_t stream) {
    const float* L = (const float*)d_in[0];
    const float* R = (const float*)d_in[1];
    float* out = (float*)d_out;

    const int B = in_sizes[0] / (C_DIM * H_DIM * W_DIM);   // 32
    dim3 grid(NDISP, H_DIM, B);
    dim3 block(128);
    size_t shmem = (size_t)2 * 64 * LSTR * sizeof(_Float16);   // 67584 B
    hipLaunchKernelGGL(corr_wmma_kernel, grid, block, shmem, stream, L, R, out);
}
